// rnn_56453050139210
// MI455X (gfx1250) — matrix operand order repaired and checked by reading, not compiled, not run on hardware
//
#include <hip/hip_runtime.h>
#include <hip/hip_bf16.h>

#define BATCH 256
#define SEQLEN 512
#define INDIM 512
#define HID 1024
#define OUTDIM 512
#define KTOT 1536              // IN + HID combined reduction
#define LDSK 1544              // padded LDS row stride (bf16 elems) -> breaks bank conflicts
#define NWG 128                // persistent grid: 16 m-tiles x 8 n-groups
#define KBLK 48                // KTOT / 32

typedef __attribute__((ext_vector_type(16))) __bf16 v16bf;
typedef __attribute__((ext_vector_type(8)))  float  v8f;

union FragU { v16bf v; uint4 q[2]; };

static __device__ __forceinline__ unsigned short f2bf_bits(float f) {
    __bf16 b = (__bf16)f;                       // RNE convert (v_cvt_*bf16_f32)
    return __builtin_bit_cast(unsigned short, b);
}

// branch-free tanh: 1 - 2/(2^(2*log2e*v) + 1); saturates correctly at +/-1
static __device__ __forceinline__ float fast_tanh(float v) {
    float e2 = __builtin_amdgcn_exp2f(v * 2.8853900817779268f);
    return 1.0f - 2.0f * __builtin_amdgcn_rcpf(e2 + 1.0f);
}

// ---------------- init: transpose/convert weights to bf16, zero h0 (bf16) & counter --------
__global__ void rnn_init_kernel(const float* __restrict__ Wxh, const float* __restrict__ Whh,
                                const float* __restrict__ fcw,
                                unsigned short* __restrict__ WcombT,   // [1024][1536] bf16
                                unsigned short* __restrict__ fcwB,     // [512][1024]  bf16 ([O][K])
                                unsigned short* __restrict__ h0,       // [256*1024] bf16
                                unsigned* __restrict__ counter) {
    size_t idx = (size_t)blockIdx.x * blockDim.x + threadIdx.x;
    size_t stride = (size_t)gridDim.x * blockDim.x;
    if (idx == 0) *counter = 0u;
    for (size_t i = idx; i < (size_t)HID * KTOT; i += stride) {
        size_t n = i / KTOT, k = i % KTOT;
        float v = (k < INDIM) ? Wxh[k * HID + n] : Whh[(k - INDIM) * HID + n];
        WcombT[i] = f2bf_bits(v);
    }
    for (size_t i = idx; i < (size_t)OUTDIM * HID; i += stride) fcwB[i] = f2bf_bits(fcw[i]);
    for (size_t i = idx; i < (size_t)BATCH * HID; i += stride) h0[i] = 0u;
}

// ---------------- device-wide barrier (monotonic counter, no per-step reset) ----------------
static __device__ __forceinline__ void grid_barrier(unsigned* counter, unsigned target) {
    __threadfence();          // release: make this thread's stores agent-visible
    __syncthreads();          // all threads of WG have fenced before rep signals
    if (threadIdx.x == 0) {
        __hip_atomic_fetch_add(counter, 1u, __ATOMIC_ACQ_REL, __HIP_MEMORY_SCOPE_AGENT);
        while (__hip_atomic_load(counter, __ATOMIC_ACQUIRE, __HIP_MEMORY_SCOPE_AGENT) < target)
            __builtin_amdgcn_s_sleep(2);
    }
    __syncthreads();
    __threadfence();          // acquire: invalidate so new h is visible to all threads
}

// ---------------- persistent recurrent kernel ----------------
__global__ void __launch_bounds__(256, 1)
rnn_persist_kernel(const float* __restrict__ x,      // [B][S][IN] f32
                   const float* __restrict__ b_h,    // [HID]
                   const float* __restrict__ fc_b,   // [OUT]
                   const unsigned short* __restrict__ WcombT, // [HID][KTOT] bf16
                   const unsigned short* __restrict__ fcwB,   // [OUT][HID]  bf16
                   unsigned short* __restrict__ hbuf,// [2][B*HID] bf16 (double buffer)
                   unsigned* __restrict__ counter,
                   float* __restrict__ y) {          // [B][OUT] f32
    __shared__ unsigned short Alds[16 * LDSK];       // 16 rows x (512 x_t | 1024 h_prev), padded

    const int tid   = threadIdx.x;
    const int wave  = tid >> 5;
    const int lane  = tid & 31;
    const int wg    = blockIdx.x;
    const int b0    = (wg >> 3) * 16;                // 16 batch rows per WG
    const int ngrp  = wg & 7;                        // 128 columns per WG, 16 per wave
    const int ncol  = (ngrp * 8 + wave) * 16 + (lane & 15);
    const int halfA = (lane >> 4) * 16;              // A frag K sub-offset (elems)
    const int halfB = (lane >> 4) * 16;              // B frag K sub-offset (elems)
    const int rbase = (lane >> 4) * 8;               // C/D local row base

    const float bh_v = b_h[ncol];
    const unsigned short* Bbase = WcombT + (size_t)ncol * KTOT + halfB;
    const unsigned short* Abase = Alds + (size_t)(lane & 15) * LDSK + halfA;

    for (int t = 0; t < SEQLEN; ++t) {
        const unsigned short* hprev = hbuf + (size_t)(t & 1) * (BATCH * HID);
        unsigned short*       hnext = hbuf + (size_t)((t + 1) & 1) * (BATCH * HID);

        // ---- stage A = [x_t (16x512 f32->bf16) | h_prev (16x1024 bf16 copy)] into LDS ----
        // 4096 groups total (8 x-iters then 8 h-iters per thread): no intra-wave divergence
        for (int g = tid; g < 4096; g += 256) {
            if (g < 2048) {                          // x_t: float4 -> 4 bf16 (8 B)
                int r = g >> 7, c = (g & 127) * 4;
                float4 f = *(const float4*)(x + ((size_t)(b0 + r) * SEQLEN + t) * INDIM + c);
                union { __bf16 h[4]; uint2 u; } p;
                p.h[0] = (__bf16)f.x; p.h[1] = (__bf16)f.y;
                p.h[2] = (__bf16)f.z; p.h[3] = (__bf16)f.w;
                *(uint2*)(&Alds[(size_t)r * LDSK + c]) = p.u;
            } else {                                 // h_prev: straight 16 B bf16 copy
                int g2 = g - 2048, r = g2 >> 7, c = (g2 & 127) * 8;
                uint4 u = *(const uint4*)(hprev + (size_t)(b0 + r) * HID + c);
                *(uint4*)(&Alds[(size_t)r * LDSK + INDIM + c]) = u;
            }
        }
        __syncthreads();

        // ---- 16x16 tile per wave, K = 1536 via 48 x v_wmma_f32_16x16x32_bf16 ----
        v8f acc;
        #pragma unroll
        for (int i = 0; i < 8; ++i) acc[i] = bh_v;

        #pragma unroll 4
        for (int kb = 0; kb < KBLK; ++kb) {
            FragU a, b;
            a.q[0] = *(const uint4*)(Abase + kb * 32);        // K: half+0..7
            a.q[1] = *(const uint4*)(Abase + kb * 32 + 8);    // K: half+8..15
            b.q[0] = *(const uint4*)(Bbase + kb * 32);        // K: half+0..15 (contig)
            b.q[1] = *(const uint4*)(Bbase + kb * 32 + 8);
            acc = __builtin_amdgcn_wmma_f32_16x16x32_bf16(
                      false, a.v, false, b.v, (short)0, acc, false, false);
        }

        // ---- h_t = tanh(acc) stored as bf16 (branch-free tanh) ----
        unsigned short* hout = hnext + (size_t)b0 * HID + ncol;
        #pragma unroll
        for (int i = 0; i < 8; ++i)
            hout[(size_t)(rbase + i) * HID] = f2bf_bits(fast_tanh(acc[i]));

        grid_barrier(counter, (unsigned)(NWG) * (unsigned)(t + 1));
    }

    // ---- y = h_final @ fc_w.T + fc_b  (h_final bf16 in hbuf[0], SEQLEN even) ----
    const unsigned short* hfin = hbuf;
    int tileId = wg * 8 + wave;                       // 1024 waves, first 512 used
    if (tileId < (BATCH / 16) * (OUTDIM / 16)) {
        int mt = tileId >> 5, nt = tileId & 31;
        int colf = nt * 16 + (lane & 15);
        const unsigned short* Bb   = fcwB + (size_t)colf * HID + halfB;
        const unsigned short* Arow = hfin + (size_t)(mt * 16 + (lane & 15)) * HID + halfA;
        float bv = fc_b[colf];
        v8f acc;
        #pragma unroll
        for (int i = 0; i < 8; ++i) acc[i] = bv;

        #pragma unroll 4
        for (int kb = 0; kb < HID / 32; ++kb) {
            FragU a, b;
            a.q[0] = *(const uint4*)(Arow + kb * 32);         // bf16 direct, no cvt
            a.q[1] = *(const uint4*)(Arow + kb * 32 + 8);
            b.q[0] = *(const uint4*)(Bb + kb * 32);
            b.q[1] = *(const uint4*)(Bb + kb * 32 + 8);
            acc = __builtin_amdgcn_wmma_f32_16x16x32_bf16(
                      false, a.v, false, b.v, (short)0, acc, false, false);
        }
        #pragma unroll
        for (int i = 0; i < 8; ++i)
            y[(size_t)(mt * 16 + rbase + i) * OUTDIM + colf] = acc[i];
    }
}

extern "C" void kernel_launch(void* const* d_in, const int* in_sizes, int n_in,
                              void* d_out, int out_size, void* d_ws, size_t ws_size,
                              hipStream_t stream) {
    const float* x    = (const float*)d_in[0];
    const float* Wxh  = (const float*)d_in[1];
    const float* Whh  = (const float*)d_in[2];
    const float* bh   = (const float*)d_in[3];
    const float* fcw  = (const float*)d_in[4];
    const float* fcb  = (const float*)d_in[5];

    // workspace carve-up (~5.3 MB)
    uintptr_t ws = (uintptr_t)d_ws;
    unsigned*       counter = (unsigned*)ws;                               // 256 B
    unsigned short* WcombT  = (unsigned short*)(ws + 256);                 // 1024*1536*2 = 3 MiB
    unsigned short* fcwB    = (unsigned short*)(ws + 256 + 3145728);       // 512*1024*2  = 1 MiB
    unsigned short* hbuf    = (unsigned short*)(ws + 256 + 3145728 + 1048576); // 2*256*1024*2 = 1 MiB

    rnn_init_kernel<<<1024, 256, 0, stream>>>(Wxh, Whh, fcw, WcombT, fcwB, hbuf, counter);
    rnn_persist_kernel<<<NWG, 256, 0, stream>>>(x, bh, fcb, WcombT, fcwB,
                                                hbuf, counter, (float*)d_out);
    (void)in_sizes; (void)n_in; (void)out_size; (void)ws_size;
}
